// CustomGRU_63058709840641
// MI455X (gfx1250) — matrix operand order repaired and checked by reading, not compiled, not run on hardware
//
#include <hip/hip_runtime.h>

typedef __attribute__((ext_vector_type(16))) _Float16 v16h;
typedef __attribute__((ext_vector_type(8)))  _Float16 v8h;
typedef __attribute__((ext_vector_type(8)))  float    v8f;

#define B_ 1024
#define T_ 4096
#define I_ 16
#define H_ 128
#define O_ 8

static __device__ __forceinline__ v16h combine16(v8h lo, v8h hi) {
    return __builtin_shufflevector(lo, hi, 0,1,2,3,4,5,6,7,8,9,10,11,12,13,14,15);
}

static __device__ __forceinline__ v8f wmma_f16(v16h a, v16h b, v8f c) {
    return __builtin_amdgcn_wmma_f32_16x16x32_f16(false, a, false, b, (short)0, c, false, false);
}

// gfx1250 V_TANH_F32 (confirmed lowering in round 2).
static __device__ __forceinline__ float fast_tanh(float v) {
#if __has_builtin(__builtin_amdgcn_tanhf)
    return __builtin_amdgcn_tanhf(v);
#else
    float e = __builtin_amdgcn_exp2f(v * -2.885390082f);
    return __builtin_amdgcn_rcpf(1.f + e) * 2.f - 1.f;
#endif
}

// Input already pre-scaled by 0.5 (weights/bias folded): sigmoid(2v) = 0.5*tanh(v)+0.5
static __device__ __forceinline__ float sigmoid_prescaled(float half_v) {
    return __builtin_fmaf(0.5f, fast_tanh(half_v), 0.5f);
}

__global__ __launch_bounds__(256, 1)
void CustomGRU_gru_kernel(const float* __restrict__ x,
                          const float* __restrict__ Wz, const float* __restrict__ bz,
                          const float* __restrict__ Wr, const float* __restrict__ br,
                          const float* __restrict__ Wh, const float* __restrict__ bh,
                          const float* __restrict__ Wo, const float* __restrict__ bo,
                          float* __restrict__ out)
{
    __shared__ alignas(16) _Float16 h_lds[16 * H_];   // 4 KB, row-major [row][k]
    __shared__ alignas(16) _Float16 rh_lds[16 * H_];  // 4 KB
    __shared__ alignas(16) float    h32[16 * H_];     // 8 KB

    const int tid  = threadIdx.x;
    const int lane = tid & 31;
    const int wave = tid >> 5;            // 0..7 : which 16-column N-tile of H
    const int b0   = (int)blockIdx.x * 16;

    const int col   = lane & 15;          // A-row / C-column-in-tile
    const int hs    = lane >> 4;          // lane-half select
    const int kbase = hs * 8;             // A-fragment K base for this lane half
    const int ncol  = wave * 16 + col;    // this lane's output column in [0,128)

    // ---------------- Preload weight B-fragments into registers ----------------
    // B layout (16-bit, 32x16): lane L holds N = L&15, K = 8*(L>=16) + (e<8 ? e : e+8), e=0..15.
    // comb = [h(0..127), x(128..143)]; chunks 0..3 cover h K=32c..32c+31,
    // chunk 4 covers x rows 128..143 zero-padded to K=32.
    // Wz/Wr (and bz/br) are pre-scaled by 0.5 so the gate is 0.5*tanh(acc)+0.5.
    v16h wzf[5], wrf[5], whf[5];
#pragma unroll
    for (int c = 0; c < 5; ++c) {
        v16h az, ar, ah;
#pragma unroll
        for (int e = 0; e < 16; ++e) {
            float vz = 0.f, vr = 0.f, vh = 0.f;
            if (c < 4) {
                int k = 32 * c + hs * 8 + (e < 8 ? e : e + 8);
                vz = 0.5f * Wz[k * H_ + ncol];
                vr = 0.5f * Wr[k * H_ + ncol];
                vh = Wh[k * H_ + ncol];
            } else if (e < 8) {           // K 16..31 of the x-chunk (e>=8) is zero pad
                int k = H_ + hs * 8 + e;
                vz = 0.5f * Wz[k * H_ + ncol];
                vr = 0.5f * Wr[k * H_ + ncol];
                vh = Wh[k * H_ + ncol];
            }
            az[e] = (_Float16)vz; ar[e] = (_Float16)vr; ah[e] = (_Float16)vh;
        }
        wzf[c] = az; wrf[c] = ar; whf[c] = ah;
    }

    const float bzi = 0.5f * bz[ncol];
    const float bri = 0.5f * br[ncol];
    const float bhi = bh[ncol];

    // h state: f32 C-layout fragment (this wave's 16 columns), plus f16 copy in LDS.
    v8f hf = {};
    for (int i = tid; i < 16 * H_; i += 256) h_lds[i] = (_Float16)0.f;
    __syncthreads();

    const float* xbase = x + (size_t)(b0 + col) * T_ * I_ + kbase;

    // Software-pipelined x tile: loads for step t issued during step t-1.
    float4 f0 = ((const float4*)xbase)[0];
    float4 f1 = ((const float4*)xbase)[1];

    for (int t = 0; t < T_; ++t) {
        // ---- build x A-fragment from the pipelined loads (K 0..15, pad to 32) ----
        v8h xlo;
        xlo[0] = (_Float16)f0.x; xlo[1] = (_Float16)f0.y;
        xlo[2] = (_Float16)f0.z; xlo[3] = (_Float16)f0.w;
        xlo[4] = (_Float16)f1.x; xlo[5] = (_Float16)f1.y;
        xlo[6] = (_Float16)f1.z; xlo[7] = (_Float16)f1.w;
        v8h zero8 = {};
        v16h ax = combine16(xlo, zero8);

        // ---- h A-fragments from LDS (two ds_load_b128 per chunk) ----
        v16h ah[4];
#pragma unroll
        for (int c = 0; c < 4; ++c) {
            const _Float16* p = &h_lds[col * H_ + 32 * c + kbase];
            v8h lo = *(const v8h*)p;
            v8h hi = *(const v8h*)(p + 16);
            ah[c] = combine16(lo, hi);
        }

        // ---- issue next step's x loads early (latency hidden behind WMMAs) ----
        {
            int tn = (t + 1 < T_) ? (t + 1) : t;
            const float* xn = xbase + (size_t)tn * I_;
            __builtin_prefetch(xn + 8 * I_, 0, 3);
            f0 = ((const float4*)xn)[0];
            f1 = ((const float4*)xn)[1];
        }

        // ---- z and r gates: 5 WMMA each, bias in C-init ----
        v8f zacc = { bzi, bzi, bzi, bzi, bzi, bzi, bzi, bzi };
        v8f racc = { bri, bri, bri, bri, bri, bri, bri, bri };
#pragma unroll
        for (int c = 0; c < 4; ++c) {
            zacc = wmma_f16(ah[c], wzf[c], zacc);
            racc = wmma_f16(ah[c], wrf[c], racc);
        }
        zacc = wmma_f16(ax, wzf[4], zacc);
        racc = wmma_f16(ax, wrf[4], racc);

        // ---- candidate x-part: independent of r*h, compute before the barrier ----
        v8f hacc = { bhi, bhi, bhi, bhi, bhi, bhi, bhi, bhi };
        hacc = wmma_f16(ax, whf[4], hacc);

        // ---- r gate + r*h -> LDS (f16) ----
#pragma unroll
        for (int j = 0; j < 8; ++j) {
            float r = sigmoid_prescaled(racc[j]);
            rh_lds[(j + 8 * hs) * H_ + ncol] = (_Float16)(r * hf[j]);
        }
        __syncthreads();

        // ---- z gate: deferred past the barrier, overlaps the r*h WMMA chain ----
        v8f zg;
#pragma unroll
        for (int j = 0; j < 8; ++j) zg[j] = sigmoid_prescaled(zacc[j]);

        // ---- candidate h-part: A = r*h from LDS ----
#pragma unroll
        for (int c = 0; c < 4; ++c) {
            const _Float16* p = &rh_lds[col * H_ + 32 * c + kbase];
            v8h lo = *(const v8h*)p;
            v8h hi = *(const v8h*)(p + 16);
            hacc = wmma_f16(combine16(lo, hi), whf[c], hacc);
        }

        // ---- h_new = h + z*(tanh(hacc) - h); publish f16 copy for next step ----
#pragma unroll
        for (int j = 0; j < 8; ++j) {
            float th = fast_tanh(hacc[j]);
            float hn = __builtin_fmaf(zg[j], th - hf[j], hf[j]);
            hf[j] = hn;
            h_lds[(j + 8 * hs) * H_ + ncol] = (_Float16)hn;
        }
        __syncthreads();
    }

    // ---------------- Output projection: [16,128] x [128,8] + bo ----------------
#pragma unroll
    for (int j = 0; j < 8; ++j) h32[(j + 8 * hs) * H_ + ncol] = hf[j];
    __syncthreads();

    if (tid < 16 * O_) {
        int r  = tid >> 3;
        int co = tid & 7;
        float acc = bo[co];
#pragma unroll 8
        for (int k = 0; k < H_; ++k) acc += h32[r * H_ + k] * Wo[k * O_ + co];
        out[(size_t)(b0 + r) * O_ + co] = acc;
    }
}

extern "C" void kernel_launch(void* const* d_in, const int* in_sizes, int n_in,
                              void* d_out, int out_size, void* d_ws, size_t ws_size,
                              hipStream_t stream) {
    const float* x  = (const float*)d_in[0];
    const float* Wz = (const float*)d_in[1];
    const float* bz = (const float*)d_in[2];
    const float* Wr = (const float*)d_in[3];
    const float* br = (const float*)d_in[4];
    const float* Wh = (const float*)d_in[5];
    const float* bh = (const float*)d_in[6];
    const float* Wo = (const float*)d_in[7];
    const float* bo = (const float*)d_in[8];
    float* out = (float*)d_out;

    dim3 grid(B_ / 16);   // 64 workgroups, one 16-row batch tile each
    dim3 block(256);      // 8 waves: one per 16-column N-tile of H
    CustomGRU_gru_kernel<<<grid, block, 0, stream>>>(x, Wz, bz, Wr, br, Wh, bh, Wo, bo, out);
}
